// RecursiveBlock_42614665511018
// MI455X (gfx1250) — matrix operand order repaired and checked by reading, not compiled, not run on hardware
//
#include <hip/hip_runtime.h>

typedef __attribute__((ext_vector_type(16))) __bf16 v16bf;
typedef __attribute__((ext_vector_type(8)))  float  v8f;
typedef __attribute__((ext_vector_type(4)))  int    v4i;

#define HW 4096   // 64*64 spatial positions per channel

union FragU { v16bf v; uint4 q[2]; };

#if defined(__has_builtin)
#  if __has_builtin(__builtin_amdgcn_global_load_async_to_lds_b128)
#    define HAVE_ASYNC_BUILTIN 1
#  endif
#  if __has_builtin(__builtin_amdgcn_s_wait_asynccnt)
#    define HAVE_WAIT_ASYNC_BUILTIN 1
#  endif
#endif

typedef v4i __attribute__((address_space(1))) v4i_g;  // global int4
typedef v4i __attribute__((address_space(3))) v4i_l;  // LDS int4

// Async 16B global->LDS copy (per-lane), tracked by ASYNCcnt.
__device__ inline void async_copy16(const float* gsrc, float* ldst) {
#if defined(HAVE_ASYNC_BUILTIN)
  __builtin_amdgcn_global_load_async_to_lds_b128(
      (v4i_g*)(unsigned long long)gsrc,
      (v4i_l*)(unsigned)(unsigned long long)ldst, 0, 0);
#else
  unsigned           l = (unsigned)(unsigned long long)ldst;  // LDS byte offset
  unsigned long long g = (unsigned long long)gsrc;
  asm volatile("global_load_async_to_lds_b128 %0, %1, off"
               :: "v"(l), "v"(g) : "memory");
#endif
}

__device__ inline void wait_async0() {
#if defined(HAVE_WAIT_ASYNC_BUILTIN)
  __builtin_amdgcn_s_wait_asynccnt(0);
#else
  asm volatile("s_wait_asynccnt 0" ::: "memory");
#endif
}

// Build an A-fragment from two contiguous 8-float runs (elements 0..7 =
// K run at half*16, elements 8..15 = K run at half*16+8), f32 -> bf16 (RNE).
__device__ inline v16bf frag_from_f32_2x8(const float* p0, const float* p1) {
  v16bf v;
#pragma unroll
  for (int i = 0; i < 8; ++i) v[i] = (__bf16)p0[i];
#pragma unroll
  for (int i = 0; i < 8; ++i) v[8 + i] = (__bf16)p1[i];
  return v;
}

// B-fragment: 16 contiguous K values of one column (element e <-> K = half*16 + e).
__device__ inline v16bf frag_from_f32_16(const float* p) {
  v16bf v;
#pragma unroll
  for (int i = 0; i < 16; ++i) v[i] = (__bf16)p[i];
  return v;
}

// ---------------------------------------------------------------------------
// 1x1-conv GEMM: Y[b][row][n] = (R? R[b][row][n] : 0) + sum_c W[row][c]*X[b][c][n]
// Block tile 64x128, K step 32, 8 waves (4 along M, 2 along N), 4 wmma/wave/step.
// A tile staged f32 via GLOBAL_LOAD_ASYNC_TO_LDS_B128; B tile staged transposed
// as packed bf16 pairs (batched loads -> single wait, b32 LDS stores).
// ---------------------------------------------------------------------------
__global__ __launch_bounds__(256) void gemm_conv1x1(
    const float* __restrict__ W, int ldw,
    const float* __restrict__ X, long xbs,
    float* __restrict__ Y, long ybs,
    const float* __restrict__ R, long rbs,
    int K)
{
  __shared__ __align__(16) float  Asf[64][36];  // 144B rows: 16B aligned, bank-spread
  __shared__ __align__(16) __bf16 Bs[128][40];  // transposed Bs[n][k], 80B rows

  const int tid  = threadIdx.x;
  const int b    = blockIdx.z;
  const int row0 = blockIdx.y * 64;
  const int n0   = blockIdx.x * 128;

  const int wv   = tid >> 5;
  const int lane = tid & 31;
  const int half = lane >> 4;
  const int lr   = lane & 15;
  const int wm   = wv & 3;   // 0..3 along M
  const int wn   = wv >> 2;  // 0..1 along N

  v8f acc[4];
#pragma unroll
  for (int t = 0; t < 4; ++t)
#pragma unroll
    for (int j = 0; j < 8; ++j) acc[t][j] = 0.0f;

  const float* Xb = X + (long)b * xbs;

  for (int k0 = 0; k0 < K; k0 += 32) {
    __syncthreads();

    // --- stage A (64x32 f32) via async DMA: 512 x 16B, 2 per thread ---
#pragma unroll
    for (int i = 0; i < 2; ++i) {
      int ti = tid + 256 * i;
      int m = ti >> 3, q = ti & 7;
      async_copy16(&W[(long)(row0 + m) * ldw + k0 + q * 4], &Asf[m][q * 4]);
    }

    // --- stage B (32x128 -> bf16 transposed): batch loads, then packed stores ---
    float b0[8], b1[8];
#pragma unroll
    for (int i = 0; i < 8; ++i) {
      int pi = tid + 256 * i;
      int n = pi & 127, kp = pi >> 7;                 // kp: 0..15 (pairs along k)
      const float* p = Xb + (long)(k0 + 2 * kp) * HW + (n0 + n);
      b0[i] = p[0];
      b1[i] = p[HW];
    }
#pragma unroll
    for (int i = 0; i < 8; ++i) {
      int pi = tid + 256 * i;
      int n = pi & 127, kp = pi >> 7;
      unsigned lo = __builtin_bit_cast(unsigned short, (__bf16)b0[i]);
      unsigned hi = __builtin_bit_cast(unsigned short, (__bf16)b1[i]);
      *reinterpret_cast<unsigned*>(&Bs[n][2 * kp]) = lo | (hi << 16);
    }

    if (k0 + 32 < K) {  // pull next K-step's X lines toward the caches
      int n = tid & 127, kp = tid >> 7;
      __builtin_prefetch(Xb + (long)(k0 + 32 + kp) * HW + (n0 + n), 0, 3);
    }

    wait_async0();
    __syncthreads();

    FragU a;
    {
      const float* arow = &Asf[wm * 16 + lr][0];
      a.v = frag_from_f32_2x8(arow + half * 16, arow + half * 16 + 8);
    }
#pragma unroll
    for (int tn = 0; tn < 4; ++tn) {
      FragU bf;
      const __bf16* brow = &Bs[wn * 64 + tn * 16 + lr][half * 16];
      bf.q[0] = reinterpret_cast<const uint4*>(brow)[0];
      bf.q[1] = reinterpret_cast<const uint4*>(brow)[1];
      acc[tn] = __builtin_amdgcn_wmma_f32_16x16x32_bf16(
          false, a.v, false, bf.v, (short)0, acc[tn], false, false);
    }
  }

  // epilogue: hoist the uniform residual branch out of the store loops
  if (R) {
#pragma unroll
    for (int tn = 0; tn < 4; ++tn)
#pragma unroll
      for (int r = 0; r < 8; ++r) {
        int gr = row0 + wm * 16 + half * 8 + r;
        int gc = n0 + wn * 64 + tn * 16 + lr;
        long off = (long)gr * HW + gc;
        Y[(long)b * ybs + off] = acc[tn][r] + R[(long)b * rbs + off];
      }
  } else {
#pragma unroll
    for (int tn = 0; tn < 4; ++tn)
#pragma unroll
      for (int r = 0; r < 8; ++r) {
        int gr = row0 + wm * 16 + half * 8 + r;
        int gc = n0 + wn * 64 + tn * 16 + lr;
        Y[(long)b * ybs + (long)gr * HW + gc] = acc[tn][r];
      }
  }
}

// ---------------------------------------------------------------------------
// L2-normalize each q/k channel row (length 4096) of the CA qkv buffer in place.
// ---------------------------------------------------------------------------
__global__ __launch_bounds__(256) void rownorm_qk(float* __restrict__ qkv)
{
  __shared__ float red[256];
  const int row = blockIdx.x;
  const int b = row >> 9, ch = row & 511;
  float* p = qkv + ((long)b * 768 + ch) * HW;
  const int tid = threadIdx.x;
  float s = 0.f;
  for (int i = tid; i < HW; i += 256) { float v = p[i]; s += v * v; }
  red[tid] = s;
  __syncthreads();
  for (int st = 128; st > 0; st >>= 1) {
    if (tid < st) red[tid] += red[tid + st];
    __syncthreads();
  }
  const float inv = 1.0f / fmaxf(sqrtf(red[0]), 1e-12f);
  for (int i = tid; i < HW; i += 256) p[i] *= inv;
}

// ---------------------------------------------------------------------------
// Channel attention for one (b, h): S = q k^T over n=4096 (WMMA, K split across
// 8 waves), reduce + *temp + row softmax in LDS, then out = P @ v (VALU, K=32).
// ---------------------------------------------------------------------------
__global__ __launch_bounds__(256) void ca_attention(
    const float* __restrict__ qkv, const float* __restrict__ temp,
    float* __restrict__ outp)
{
  __shared__ float Sp[8][32][32];
  __shared__ float S[32][32];

  const int bh = blockIdx.x;
  const int b = bh >> 3, h = bh & 7;
  const float* qn = qkv + ((long)b * 768 + h * 32) * HW;
  const float* kn = qkv + ((long)b * 768 + 256 + h * 32) * HW;
  const float* vn = qkv + ((long)b * 768 + 512 + h * 32) * HW;

  const int tid = threadIdx.x;
  const int wv = tid >> 5, lane = tid & 31, half = lane >> 4, lr = lane & 15;

  v8f acc[2][2];
#pragma unroll
  for (int mb = 0; mb < 2; ++mb)
#pragma unroll
    for (int nb = 0; nb < 2; ++nb)
#pragma unroll
      for (int j = 0; j < 8; ++j) acc[mb][nb][j] = 0.0f;

  for (int step = 0; step < 16; ++step) {
    const int k0 = wv * 512 + step * 32;
    FragU a[2], bb[2];
#pragma unroll
    for (int mb = 0; mb < 2; ++mb) {
      const float* p = qn + (long)(mb * 16 + lr) * HW + k0 + half * 16;
      a[mb].v = frag_from_f32_2x8(p, p + 8);
    }
#pragma unroll
    for (int nb = 0; nb < 2; ++nb) {
      const float* p = kn + (long)(nb * 16 + lr) * HW + k0 + half * 16;
      bb[nb].v = frag_from_f32_16(p);
    }
#pragma unroll
    for (int mb = 0; mb < 2; ++mb)
#pragma unroll
      for (int nb = 0; nb < 2; ++nb)
        acc[mb][nb] = __builtin_amdgcn_wmma_f32_16x16x32_bf16(
            false, a[mb].v, false, bb[nb].v, (short)0, acc[mb][nb], false, false);
  }

#pragma unroll
  for (int mb = 0; mb < 2; ++mb)
#pragma unroll
    for (int nb = 0; nb < 2; ++nb)
#pragma unroll
      for (int r = 0; r < 8; ++r)
        Sp[wv][mb * 16 + half * 8 + r][nb * 16 + lr] = acc[mb][nb][r];
  __syncthreads();

  const float tval = temp[h];
  for (int e = tid; e < 1024; e += 256) {
    int c = e >> 5, d = e & 31;
    float s = 0.f;
#pragma unroll
    for (int w = 0; w < 8; ++w) s += Sp[w][c][d];
    S[c][d] = s * tval;
  }
  __syncthreads();

  if (tid < 32) {
    float m = -3.0e38f;
    for (int d = 0; d < 32; ++d) m = fmaxf(m, S[tid][d]);
    float sum = 0.f;
    for (int d = 0; d < 32; ++d) { float e = __expf(S[tid][d] - m); S[tid][d] = e; sum += e; }
    float inv = 1.f / sum;
    for (int d = 0; d < 32; ++d) S[tid][d] *= inv;
  }
  __syncthreads();

  float* ob = outp + ((long)b * 256 + h * 32) * HW;
  for (int it = 0; it < 16; ++it) {
    const int n = tid + it * 256;
    float vcol[32];
#pragma unroll
    for (int d = 0; d < 32; ++d) vcol[d] = vn[(long)d * HW + n];
#pragma unroll
    for (int c = 0; c < 32; ++c) {
      float s = 0.f;
#pragma unroll
      for (int d = 0; d < 32; ++d) s += S[c][d] * vcol[d];
      ob[(long)c * HW + n] = s;
    }
  }
}

// ---------------------------------------------------------------------------
// CSWin window attention (flash-style, one token per thread, 8 heads looped).
// Horizontal: windows (b, yb) of 64x4; Vertical: windows (b, xb) of 4x64.
// qkv2 layout: [b][384][4096]  (q:0..127, k:128..255, v:256..383).
// ---------------------------------------------------------------------------
__global__ __launch_bounds__(256) void cswin_attention(
    const float* __restrict__ qkv2, const float* __restrict__ lepe_v,
    float* __restrict__ outp, int vertical)
{
  __shared__ float Ksh[256][8];
  __shared__ float Vsh[256][8];

  const int b = blockIdx.x >> 4;
  const int wi = blockIdx.x & 15;
  const int t = threadIdx.x;
  int n;
  if (!vertical) { int i1 = t >> 2, i2 = t & 3;  n = i1 * 64 + wi * 4 + i2; }
  else           { int i1 = t >> 6, i2 = t & 63; n = (wi * 4 + i1) * 64 + i2; }

  const float* base = qkv2 + (long)b * 384 * HW;
  const float scale = 0.25f;  // (128/8)^-0.5

  for (int h = 0; h < 8; ++h) {
    const int qc = (vertical ? 64 : 0) + h * 8;
    const int kc = 128 + qc;
    const int vc = 256 + qc;
    __syncthreads();
    float q[8];
#pragma unroll
    for (int j = 0; j < 8; ++j) {
      q[j] = base[(long)(qc + j) * HW + n] * scale;
      Ksh[t][j] = base[(long)(kc + j) * HW + n];
      Vsh[t][j] = base[(long)(vc + j) * HW + n];
    }
    __syncthreads();

    float m = -3.0e38f, l = 0.f, o[8];
#pragma unroll
    for (int j = 0; j < 8; ++j) o[j] = 0.f;
    for (int k = 0; k < 256; ++k) {
      float s = 0.f;
#pragma unroll
      for (int j = 0; j < 8; ++j) s += q[j] * Ksh[k][j];
      float mn = fmaxf(m, s);
      float cf = __expf(m - mn);
      float p  = __expf(s - mn);
      l = l * cf + p;
#pragma unroll
      for (int j = 0; j < 8; ++j) o[j] = o[j] * cf + p * Vsh[k][j];
      m = mn;
    }
    const float inv = 1.f / l;
    float* op = outp + (long)b * 128 * HW;
#pragma unroll
    for (int j = 0; j < 8; ++j) {
      float lep = vertical ? lepe_v[((long)b * 64 + (h * 8 + j)) * HW + n]
                           : Vsh[t][j];  // horizontal lepe = v_h itself
      op[(long)(qc + j) * HW + n] = o[j] * inv + lep;
    }
  }
}

extern "C" void kernel_launch(void* const* d_in, const int* in_sizes, int n_in,
                              void* d_out, int out_size, void* d_ws, size_t ws_size,
                              hipStream_t stream)
{
  const float* x         = (const float*)d_in[0];
  const float* ca_temp   = (const float*)d_in[1];
  const float* ca_qkv_w  = (const float*)d_in[2];  // (768,128)
  const float* ca_proj_w = (const float*)d_in[3];  // (128,256)
  const float* cs_qk_w   = (const float*)d_in[4];  // (256,128)
  const float* cs_v_w    = (const float*)d_in[5];  // (128,128)
  const float* cs_vv_w   = (const float*)d_in[6];  // (64,64)
  // d_in[7] = cs_vh_w: unused by the reference
  const float* cs_proj_w = (const float*)d_in[8];  // (128,128)
  float* out = (float*)d_out;

  float* ws = (float*)d_ws;
  const long NB = 16;
  float* xa    = ws;                      // 16*128*4096  (post-CA feature map)
  float* big   = xa + NB * 128L * HW;     // 16*768*4096  (CA qkv; reused in stage B)
  float* caout = big + NB * 768L * HW;    // 16*256*4096  (CA attention output)
  float* qkv2  = big;                     // 16*384*4096  (cswin q|k|v, reuse)
  float* lepe  = big + NB * 384L * HW;    // 16*64*4096   (vertical lepe, reuse)
  float* csout = caout;                   // 16*128*4096  (cswin attn out, reuse)

  dim3 blk(256);

  // ---- Stage A: channel attention ----
  gemm_conv1x1<<<dim3(32, 12, 16), blk, 0, stream>>>(
      ca_qkv_w, 128, x, 128L * HW, big, 768L * HW, nullptr, 0, 128);
  rownorm_qk<<<dim3(16 * 512), blk, 0, stream>>>(big);
  ca_attention<<<dim3(128), blk, 0, stream>>>(big, ca_temp, caout);
  gemm_conv1x1<<<dim3(32, 2, 16), blk, 0, stream>>>(
      ca_proj_w, 256, caout, 256L * HW, xa, 128L * HW, x, 128L * HW, 256);

  // ---- Stage B: CSWin ----
  gemm_conv1x1<<<dim3(32, 4, 16), blk, 0, stream>>>(
      cs_qk_w, 128, xa, 128L * HW, qkv2, 384L * HW, nullptr, 0, 128);
  gemm_conv1x1<<<dim3(32, 2, 16), blk, 0, stream>>>(
      cs_v_w, 128, xa, 128L * HW, qkv2 + 256L * HW, 384L * HW, nullptr, 0, 128);
  gemm_conv1x1<<<dim3(32, 1, 16), blk, 0, stream>>>(
      cs_vv_w, 64, qkv2 + 320L * HW, 384L * HW, lepe, 64L * HW, nullptr, 0, 64);
  cswin_attention<<<dim3(256), blk, 0, stream>>>(qkv2, lepe, csout, 0);
  cswin_attention<<<dim3(256), blk, 0, stream>>>(qkv2, lepe, csout, 1);
  gemm_conv1x1<<<dim3(32, 2, 16), blk, 0, stream>>>(
      cs_proj_w, 128, csout, 128L * HW, out, 128L * HW, xa, 128L * HW, 128);
}
